// Convpass_hypernet_mask_22359599743420
// MI455X (gfx1250) — compile-verified, matrix operand order checked by reading, not run on hardware
//
#include <hip/hip_runtime.h>
#include <hip/hip_bf16.h>
#include <stddef.h>

// ---------------------------------------------------------------------------
// Convpass hypernet mask block for MI455X (gfx1250), bf16 WMMA + f32 accum.
// Weights pre-swizzled into WMMA B-fragment "blob" layout so fragments load
// as b128/b32 packed data instead of per-element gathers.
// ---------------------------------------------------------------------------

typedef __attribute__((ext_vector_type(16))) __bf16 v16bf;
typedef __attribute__((ext_vector_type(8)))  float  v8f;
typedef __attribute__((ext_vector_type(4)))  int    i4vec;
typedef unsigned int uint32;

#define NB 64
#define LL 577
#define CC 768
#define DIM 32
#define NMASK 5
#define PWH 24
#define H1 64
#define TOK (NB * LL)          // 36928
#define PTL (LL - 1)           // 576
#define MTILES (TOK / 16)      // 2308

#if defined(__gfx1250__) && __has_builtin(__builtin_amdgcn_global_load_async_to_lds_b128) && __has_builtin(__builtin_amdgcn_s_wait_asynccnt)
#define HAVE_ASYNC_LDS 1
typedef __attribute__((address_space(1))) i4vec GI4;   // global int4
typedef __attribute__((address_space(3))) i4vec LI4;   // LDS int4
#else
#define HAVE_ASYNC_LDS 0
#endif

union UFrag {
  v16bf v;
  uint32 u[8];
  uint4  q[2];
};

__device__ __forceinline__ v8f vzero8() {
  v8f z;
#pragma unroll
  for (int i = 0; i < 8; ++i) z[i] = 0.0f;
  return z;
}

// K of element e (0..15) in a 16x32 bf16 fragment: K = (v>>2)*16+half*8+(v&3)*2+(e&1)
__device__ __forceinline__ int kmapA(int e, int half) {
  int v = e >> 1;
  return ((v >> 2) << 4) | (half << 3) | ((v & 3) << 1) | (e & 1);
}
// even K base of dword v (pair e=2v,2v+1)
__device__ __forceinline__ int kpair(int v, int half) {
  return ((v >> 2) << 4) | (half << 3) | ((v & 3) << 1);
}

__device__ __forceinline__ v8f wmma_bf16(v16bf a, v16bf b, v8f c) {
  return __builtin_amdgcn_wmma_f32_16x16x32_bf16(
      false, a, false, b, (short)0, c, false, false);
}

__device__ __forceinline__ float qgelu(float v) {
  return v / (1.0f + __expf(-1.702f * v));
}

__device__ __forceinline__ uint32 pack2bf(float lo, float hi) {
  union { __bf16 b; unsigned short s; } a, c;
  a.b = (__bf16)lo;
  c.b = (__bf16)hi;
  return (uint32)a.s | ((uint32)c.s << 16);
}

// Load a pre-swizzled B fragment blob (1024 B): two b128 per lane.
__device__ __forceinline__ UFrag load_blob(const __bf16* base, int blob, int lane) {
  UFrag f;
  const uint4* p = (const uint4*)base + blob * 64 + lane * 2;
  f.q[0] = p[0];
  f.q[1] = p[1];
  return f;
}

// ---------------------------------------------------------------------------
// Kernel 0: pack W1/W2/Wd/Wu/Wh (f32) into bf16 B-fragment blob layout.
// dst element index: blob*512 + lane*16 + e ; blob = (k0/32)*(N/16) + nt
// ---------------------------------------------------------------------------
__device__ __forceinline__ void pack_one(__bf16* dst, const float* src,
                                         int N, int idx) {
  int e = idx & 15;
  int lane = (idx >> 4) & 31;
  int blob = idx >> 9;
  int ntiles = N >> 4;
  int kb = blob / ntiles, nt = blob % ntiles;
  int half = lane >> 4;
  int K = kb * 32 + kmapA(e, half);
  int n = nt * 16 + (lane & 15);
  dst[idx] = (__bf16)src[(size_t)K * N + n];
}

__global__ void k0_pack(const float* __restrict__ W1, const float* __restrict__ W2,
                        const float* __restrict__ Wd, const float* __restrict__ Wu,
                        const float* __restrict__ Wh,
                        __bf16* w1p, __bf16* w2p, __bf16* wdp, __bf16* wup,
                        __bf16* whp) {
  int i = blockIdx.x * 256 + threadIdx.x;
  if (i < 49152)        pack_one(w1p, W1, H1, i);               // 768x64
  else if (i < 53248)   pack_one(w2p, W2, H1, i - 49152);       // 64x64
  else if (i < 77824)   pack_one(wdp, Wd, DIM, i - 53248);      // 768x32
  else if (i < 102400)  pack_one(wup, Wu, CC, i - 77824);       // 32x768
  else if (i < 692224)  pack_one(whp, Wh, 9216, i - 102400);    // 64x9216
}

// ---------------------------------------------------------------------------
// Kernel 1: fused  prompt = relu(x@W1+b1)@W2+b2  and  xd = qgelu(x@Wd+bd)
// One wave per 16-token tile, 8 waves/block.
// ---------------------------------------------------------------------------
__global__ __launch_bounds__(256) void k1_proj(
    const float* __restrict__ x, const float* __restrict__ b1,
    const float* __restrict__ b2, const float* __restrict__ bd,
    const __bf16* __restrict__ w1p, const __bf16* __restrict__ w2p,
    const __bf16* __restrict__ wdp,
    float* __restrict__ prompt, float* __restrict__ xdf, __bf16* __restrict__ xdb) {
  const int tid = threadIdx.x;
  const int wid = tid >> 5;
  const int lane = tid & 31;
  const int half = lane >> 4;
  const int l15 = lane & 15;
  const int mt = blockIdx.x * 8 + wid;
  __shared__ __bf16 stage[8][16 * H1];   // per-wave relayout buffer (16 KB)
  if (mt >= MTILES) return;
  const int m0 = mt * 16;

  v8f acc1[4], accd[2];
#pragma unroll
  for (int i = 0; i < 4; ++i) acc1[i] = vzero8();
#pragma unroll
  for (int i = 0; i < 2; ++i) accd[i] = vzero8();

  const float* xrow = x + (size_t)(m0 + l15) * CC;
  for (int k0 = 0; k0 < CC; k0 += 32) {
    const int kb = k0 >> 5;
    // A: 8 x float2 loads + packed bf16 converts
    UFrag a;
#pragma unroll
    for (int v = 0; v < 8; ++v) {
      float2 f2 = *(const float2*)(xrow + k0 + kpair(v, half));
      a.u[v] = pack2bf(f2.x, f2.y);
    }
    // hoist all 6 B fragments before issuing the 6 WMMAs (clause the loads)
    UFrag bw[6];
#pragma unroll
    for (int nt = 0; nt < 4; ++nt) bw[nt] = load_blob(w1p, kb * 4 + nt, lane);
#pragma unroll
    for (int nt = 0; nt < 2; ++nt) bw[4 + nt] = load_blob(wdp, kb * 2 + nt, lane);
#pragma unroll
    for (int nt = 0; nt < 4; ++nt) acc1[nt] = wmma_bf16(a.v, bw[nt].v, acc1[nt]);
#pragma unroll
    for (int nt = 0; nt < 2; ++nt) accd[nt] = wmma_bf16(a.v, bw[4 + nt].v, accd[nt]);
  }

  // bias + relu -> bf16 LDS stage (C-frag -> A-frag relayout, in-wave order)
  __bf16* stg = stage[wid];
#pragma unroll
  for (int nt = 0; nt < 4; ++nt) {
    int col = nt * 16 + l15;
    float bias = b1[col];
#pragma unroll
    for (int r = 0; r < 8; ++r)
      stg[(r + (half << 3)) * H1 + col] = (__bf16)fmaxf(acc1[nt][r] + bias, 0.0f);
  }

  // second GEMM: (16 x 64) @ W2 (64 x 64)
  v8f accp[4];
#pragma unroll
  for (int i = 0; i < 4; ++i) accp[i] = vzero8();
#pragma unroll
  for (int ks = 0; ks < H1; ks += 32) {
    UFrag a2;
#pragma unroll
    for (int v = 0; v < 8; ++v)
      a2.u[v] = *(const uint32*)(stg + l15 * H1 + ks + kpair(v, half));
    UFrag bw[4];
#pragma unroll
    for (int nt = 0; nt < 4; ++nt) bw[nt] = load_blob(w2p, (ks >> 5) * 4 + nt, lane);
#pragma unroll
    for (int nt = 0; nt < 4; ++nt) accp[nt] = wmma_bf16(a2.v, bw[nt].v, accp[nt]);
  }
#pragma unroll
  for (int nt = 0; nt < 4; ++nt) {
    int col = nt * 16 + l15;
    float bias = b2[col];
#pragma unroll
    for (int r = 0; r < 8; ++r)
      prompt[(size_t)(m0 + r + (half << 3)) * H1 + col] = accp[nt][r] + bias;
  }
#pragma unroll
  for (int nt = 0; nt < 2; ++nt) {
    int col = nt * 16 + l15;
    float bias = bd[col];
#pragma unroll
    for (int r = 0; r < 8; ++r) {
      float g = qgelu(accd[nt][r] + bias);
      size_t idx = (size_t)(m0 + r + (half << 3)) * DIM + col;
      xdf[idx] = g;
      xdb[idx] = (__bf16)g;
    }
  }
}

// ---------------------------------------------------------------------------
// Kernel 2: mask logits + softmax over L + msum + feat (one block per batch)
// ---------------------------------------------------------------------------
__global__ __launch_bounds__(256) void k2_mask(
    const float* __restrict__ prompt, const float* __restrict__ mask_w,
    float* __restrict__ featf, __bf16* __restrict__ featb) {
  const int b = blockIdx.x, tid = threadIdx.x;
  __shared__ float mw[NMASK][H1];
  __shared__ float logit[NMASK][PTL];
  __shared__ float red[256];
  __shared__ float stat[NMASK][2];
  __shared__ float msum_s[PTL];
  __shared__ float part[H1][4];

  for (int i = tid; i < NMASK * H1; i += 256) mw[i / H1][i % H1] = mask_w[i];
  __syncthreads();

  const float* ptb = prompt + ((size_t)b * LL + 1) * H1;
  for (int l = tid; l < PTL; l += 256) {
    const float* row = ptb + (size_t)l * H1;
    float d[NMASK] = {0, 0, 0, 0, 0};
    for (int k = 0; k < H1; ++k) {
      float xv = row[k];
#pragma unroll
      for (int m = 0; m < NMASK; ++m) d[m] += xv * mw[m][k];
    }
#pragma unroll
    for (int m = 0; m < NMASK; ++m) logit[m][l] = d[m];
  }
  __syncthreads();

  for (int m = 0; m < NMASK; ++m) {
    float lm = -3.0e38f;
    for (int l = tid; l < PTL; l += 256) lm = fmaxf(lm, logit[m][l]);
    red[tid] = lm; __syncthreads();
    for (int s = 128; s > 0; s >>= 1) {
      if (tid < s) red[tid] = fmaxf(red[tid], red[tid + s]);
      __syncthreads();
    }
    if (tid == 0) stat[m][0] = red[0];
    __syncthreads();
    float mx = stat[m][0], ls = 0.0f;
    for (int l = tid; l < PTL; l += 256) ls += __expf(logit[m][l] - mx);
    red[tid] = ls; __syncthreads();
    for (int s = 128; s > 0; s >>= 1) {
      if (tid < s) red[tid] += red[tid + s];
      __syncthreads();
    }
    if (tid == 0) stat[m][1] = red[0];
    __syncthreads();
  }

  for (int l = tid; l < PTL; l += 256) {
    float s = 0.0f;
#pragma unroll
    for (int m = 0; m < NMASK; ++m)
      s += __expf(logit[m][l] - stat[m][0]) / stat[m][1];
    msum_s[l] = s;
  }
  __syncthreads();

  const int k = tid & 63, g = tid >> 6;
  float acc = 0.0f;
  for (int dl = 0; dl < 144; ++dl) {
    int l = g * 144 + dl;
    acc += msum_s[l] * ptb[(size_t)l * H1 + k];
  }
  part[k][g] = acc;
  __syncthreads();
  if (tid < H1) {
    float f = part[tid][0] + part[tid][1] + part[tid][2] + part[tid][3];
    featf[b * H1 + tid] = f;
    featb[b * H1 + tid] = (__bf16)f;
  }
}

// ---------------------------------------------------------------------------
// Kernel 3: conv_w = feat @ Wh + bh  (M=64, N=9216, K=64) via WMMA
// ---------------------------------------------------------------------------
__global__ __launch_bounds__(256) void k3_hyper(
    const __bf16* __restrict__ featb, const __bf16* __restrict__ whp,
    const float* __restrict__ bh, float* __restrict__ cwf, __bf16* __restrict__ cwb) {
  const int tid = threadIdx.x, wid = tid >> 5, lane = tid & 31;
  const int half = lane >> 4, l15 = lane & 15;
  const int job = blockIdx.x * 8 + wid;     // 2304 jobs
  const int mt = job / 576, nt = job % 576;

  UFrag a0, a1;
#pragma unroll
  for (int v = 0; v < 8; ++v) {
    const __bf16* fr = featb + (size_t)(mt * 16 + l15) * H1;
    a0.u[v] = *(const uint32*)(fr + kpair(v, half));
    a1.u[v] = *(const uint32*)(fr + 32 + kpair(v, half));
  }
  UFrag b0 = load_blob(whp, nt, lane);
  UFrag b1 = load_blob(whp, 576 + nt, lane);
  v8f acc = wmma_bf16(a0.v, b0.v, vzero8());
  acc = wmma_bf16(a1.v, b1.v, acc);

  int col = nt * 16 + l15;
  float bias = bh[col];
#pragma unroll
  for (int r = 0; r < 8; ++r) {
    int row = mt * 16 + r + (half << 3);
    float v = acc[r] + bias;
    size_t idx = (size_t)row * 9216 + col;
    cwf[idx] = v;
    cwb[idx] = (__bf16)v;
  }
}

// ---------------------------------------------------------------------------
// Kernel 4: per-batch 3x3 conv as im2col GEMM (576x288x32) + qgelu + @Wu.
// xd tile + blob-packed conv weights + blob-packed Wu staged in LDS.
// Contiguous fills use GLOBAL_LOAD_ASYNC_TO_LDS when the toolchain has it.
// ---------------------------------------------------------------------------
__global__ __launch_bounds__(256) void k4_conv(
    const __bf16* __restrict__ xdb, const __bf16* __restrict__ cwb,
    const __bf16* __restrict__ wup, const float* __restrict__ bu,
    float* __restrict__ out) {
  extern __shared__ char smem[];
  __bf16* xs  = (__bf16*)smem;                     // 576*32       (36864 B)
  __bf16* wc  = xs + PTL * DIM;                    // 18 blobs     (18432 B)
  __bf16* wus = wc + 288 * DIM;                    // 48 blobs     (49152 B)
  __bf16* stgall = wus + DIM * CC;                 // 8 * 16*32    ( 8192 B)

  const int b = blockIdx.x, tid = threadIdx.x;
  const int wid = tid >> 5, lane = tid & 31, half = lane >> 4, l15 = lane & 15;
  __bf16* stg = stgall + wid * (16 * DIM);

  // ---- cooperative fills ----
  {
    const char* gx = (const char*)(xdb + ((size_t)b * LL + 1) * DIM);   // 36864 B
    const char* gw = (const char*)wup;                                  // 49152 B
#if HAVE_ASYNC_LDS
    for (int i = tid * 16; i < PTL * DIM * 2; i += 256 * 16)
      __builtin_amdgcn_global_load_async_to_lds_b128(
          (GI4*)(gx + i), (LI4*)((char*)xs + i), 0, 0);
    for (int i = tid * 16; i < DIM * CC * 2; i += 256 * 16)
      __builtin_amdgcn_global_load_async_to_lds_b128(
          (GI4*)(gw + i), (LI4*)((char*)wus + i), 0, 0);
#else
    {
      const uint4* s0 = (const uint4*)gx;
      uint4* d0 = (uint4*)xs;
      for (int i = tid; i < PTL * DIM * 2 / 16; i += 256) d0[i] = s0[i];
      const uint4* s1 = (const uint4*)gw;
      uint4* d1 = (uint4*)wus;
      for (int i = tid; i < DIM * CC * 2 / 16; i += 256) d1[i] = s1[i];
    }
#endif
    // conv weights: plain (b,o,i,kh,kw) -> blob layout of Wc[k=(kh*3+kw)*32+i][o]
    const __bf16* cw = cwb + (size_t)b * 9216;
    for (int idx = tid; idx < 9216; idx += 256) {
      int e = idx & 15;
      int ln = (idx >> 4) & 31;
      int blob = idx >> 9;                 // 18 blobs: kb*2 + nt
      int kb = blob >> 1, nt = blob & 1;
      int hf = ln >> 4;
      int K = kb * 32 + kmapA(e, hf);      // K = t*32 + i2  with t = kb
      int i2 = K & 31;
      int o = nt * 16 + (ln & 15);
      wc[idx] = cw[o * 288 + i2 * 9 + kb];
    }
#if HAVE_ASYNC_LDS
    __builtin_amdgcn_s_wait_asynccnt(0);
#endif
  }
  __syncthreads();

  for (int t16 = wid; t16 < 36; t16 += 8) {
    v8f acc[2];
    acc[0] = vzero8(); acc[1] = vzero8();
    const int lrow = t16 * 16 + l15;
    const int h = lrow / PWH, w = lrow % PWH;

    for (int kk = 0; kk < 9; ++kk) {       // K = 288 = 9 * 32
      UFrag a;
#pragma unroll
      for (int v = 0; v < 8; ++v) {
        int K = kk * 32 + kpair(v, half);  // even: pair shares spatial tap
        int kh = K / 96, rem = K % 96, kw2 = rem >> 5, i2 = rem & 31;
        int hh = h + kh - 1, ww = w + kw2 - 1;
        uint32 val = 0u;
        if (hh >= 0 && hh < PWH && ww >= 0 && ww < PWH)
          val = *(const uint32*)(xs + ((hh * PWH + ww) << 5) + i2);
        a.u[v] = val;
      }
      UFrag b0 = load_blob(wc, kk * 2 + 0, lane);
      UFrag b1 = load_blob(wc, kk * 2 + 1, lane);
      acc[0] = wmma_bf16(a.v, b0.v, acc[0]);
      acc[1] = wmma_bf16(a.v, b1.v, acc[1]);
    }

    // qgelu -> bf16 stage -> A fragment (K = 32)
#pragma unroll
    for (int nt = 0; nt < 2; ++nt)
#pragma unroll
      for (int r = 0; r < 8; ++r)
        stg[(r + (half << 3)) * DIM + nt * 16 + l15] = (__bf16)qgelu(acc[nt][r]);

    UFrag ay;
#pragma unroll
    for (int v = 0; v < 8; ++v)
      ay.u[v] = *(const uint32*)(stg + l15 * DIM + kpair(v, half));

    const size_t orow = (size_t)b * LL + 1 + t16 * 16;
    for (int nt0 = 0; nt0 < 48; nt0 += 4) {  // N = 768, K = 32; 4 tiles/group
      UFrag bb[4];
#pragma unroll
      for (int j = 0; j < 4; ++j) bb[j] = load_blob(wus, nt0 + j, lane);
      v8f o8[4];
#pragma unroll
      for (int j = 0; j < 4; ++j) o8[j] = wmma_bf16(ay.v, bb[j].v, vzero8());
#pragma unroll
      for (int j = 0; j < 4; ++j) {
        int col = (nt0 + j) * 16 + l15;
        float bias = bu[col];
#pragma unroll
        for (int r = 0; r < 8; ++r)
          out[(orow + r + (half << 3)) * CC + col] = o8[j][r] + bias;
      }
    }
  }
}

// ---------------------------------------------------------------------------
// Kernel 5: CLS row:  qgelu(center-tap einsum) @ Wu + bu
// ---------------------------------------------------------------------------
__global__ __launch_bounds__(256) void k5_cls(
    const float* __restrict__ xdf, const float* __restrict__ cwf,
    const float* __restrict__ Wu, const float* __restrict__ bu,
    float* __restrict__ out) {
  const int b = blockIdx.x, tid = threadIdx.x;
  __shared__ float y0[DIM];
  if (tid < DIM) {
    const float* x0 = xdf + (size_t)b * LL * DIM;
    const float* cwrow = cwf + (size_t)b * 9216 + tid * 288;
    float s = 0.0f;
    for (int i = 0; i < DIM; ++i) s += x0[i] * cwrow[i * 9 + 4];
    y0[tid] = qgelu(s);
  }
  __syncthreads();
  for (int n = tid; n < CC; n += 256) {
    float s = bu[n];
    for (int o = 0; o < DIM; ++o) s += y0[o] * Wu[o * CC + n];
    out[(size_t)b * LL * CC + n] = s;
  }
}

// ---------------------------------------------------------------------------
// Host launch
// ---------------------------------------------------------------------------
extern "C" void kernel_launch(void* const* d_in, const int* in_sizes, int n_in,
                              void* d_out, int out_size, void* d_ws, size_t ws_size,
                              hipStream_t stream) {
  const float* x      = (const float*)d_in[0];
  const float* W1     = (const float*)d_in[1];
  const float* b1     = (const float*)d_in[2];
  const float* W2     = (const float*)d_in[3];
  const float* b2     = (const float*)d_in[4];
  const float* mask_w = (const float*)d_in[5];
  const float* Wd     = (const float*)d_in[6];
  const float* bd     = (const float*)d_in[7];
  const float* Wu     = (const float*)d_in[8];
  const float* bu     = (const float*)d_in[9];
  const float* Wh     = (const float*)d_in[10];
  const float* bh     = (const float*)d_in[11];
  float* out = (float*)d_out;

  char* ws = (char*)d_ws;
  size_t off = 0;
  float*  prompt = (float*)(ws + off);  off += (size_t)TOK * H1 * 4;
  float*  xdf    = (float*)(ws + off);  off += (size_t)TOK * DIM * 4;
  __bf16* xdb    = (__bf16*)(ws + off); off += (size_t)TOK * DIM * 2;
  float*  featf  = (float*)(ws + off);  off += (size_t)NB * H1 * 4;
  __bf16* featb  = (__bf16*)(ws + off); off += (size_t)NB * H1 * 2;
  float*  cwf    = (float*)(ws + off);  off += (size_t)NB * 9216 * 4;
  __bf16* cwb    = (__bf16*)(ws + off); off += (size_t)NB * 9216 * 2;
  __bf16* w1p    = (__bf16*)(ws + off); off += (size_t)CC * H1 * 2;
  __bf16* w2p    = (__bf16*)(ws + off); off += (size_t)H1 * H1 * 2;
  __bf16* wdp    = (__bf16*)(ws + off); off += (size_t)CC * DIM * 2;
  __bf16* wup    = (__bf16*)(ws + off); off += (size_t)DIM * CC * 2;
  __bf16* whp    = (__bf16*)(ws + off); off += (size_t)H1 * 9216 * 2;
  (void)ws_size; (void)in_sizes; (void)n_in; (void)out_size;

  // K0: pack all weights into WMMA B-fragment blob layout (bf16)
  k0_pack<<<(692224 + 255) / 256, 256, 0, stream>>>(W1, W2, Wd, Wu, Wh,
                                                    w1p, w2p, wdp, wup, whp);

  // K1: fused prompt + xd projection
  k1_proj<<<(MTILES + 7) / 8, 256, 0, stream>>>(x, b1, b2, bd, w1p, w2p, wdp,
                                                prompt, xdf, xdb);

  // K2: per-batch mask softmax + feat
  k2_mask<<<NB, 256, 0, stream>>>(prompt, mask_w, featf, featb);

  // K3: conv_w hypernetwork GEMM
  k3_hyper<<<2304 / 8, 256, 0, stream>>>(featb, whp, bh, cwf, cwb);

  // K4: per-batch conv-as-GEMM + qgelu + @Wu
  const size_t k4_smem = (size_t)(PTL * DIM + 288 * DIM + DIM * CC) * 2
                       + (size_t)8 * 16 * DIM * 2;   // 112,640 B
  (void)hipFuncSetAttribute((const void*)k4_conv,
                            hipFuncAttributeMaxDynamicSharedMemorySize,
                            (int)k4_smem);
  k4_conv<<<NB, 256, k4_smem, stream>>>(xdb, cwb, wup, bu, out);

  // K5: CLS rows
  k5_cls<<<NB, 256, 0, stream>>>(xdf, cwf, Wu, bu, out);
}
